// MemoryAttention_10118942950052
// MI455X (gfx1250) — matrix operand order repaired and checked by reading, not compiled, not run on hardware
//
#include <hip/hip_runtime.h>
#include <hip/hip_bf16.h>

typedef __attribute__((ext_vector_type(16))) _Float16 v16h;
typedef __attribute__((ext_vector_type(4)))  _Float16 v4h;
typedef __attribute__((ext_vector_type(8)))  float    v8f;

#define WMMA_F32_F16(A, B, C) \
  __builtin_amdgcn_wmma_f32_16x16x32_f16(false, (A), false, (B), (short)0, (C), false, false)

// ---- CDNA5 async global->LDS copy (ASYNCcnt) with safe fallbacks -----------
#if defined(__has_builtin)
# if __has_builtin(__builtin_amdgcn_global_load_async_to_lds_b128)
#  define USE_ASYNC_LDS 1
# endif
#endif

#if defined(USE_ASYNC_LDS)
# if __has_builtin(__builtin_amdgcn_s_wait_asynccnt)
#  define WAIT_ASYNC(n) __builtin_amdgcn_s_wait_asynccnt(n)
# else
#  define WAIT_ASYNC(n) asm volatile("s_wait_asynccnt %0" ::"i"(n) : "memory")
# endif
#else
# define WAIT_ASYNC(n)
#endif

// Builtin signature (from clang diagnostic): v4i addrspace(1)* src,
// v4i addrspace(3)* dst, imm offset, imm cpol.
typedef int v4i_vs __attribute__((vector_size(16)));
typedef __attribute__((address_space(1))) v4i_vs* g1_v4i;
typedef __attribute__((address_space(3))) v4i_vs* l3_v4i;

// Copy one 8KB tile (64x64 f16) global -> LDS.
// Async path: 4 x b128 per thread (128 threads), tracked by ASYNCcnt.
__device__ __forceinline__ void tile_copy_8k(const _Float16* __restrict__ gsrc,
                                             _Float16* __restrict__ lds) {
#if defined(USE_ASYNC_LDS)
  const int tid = threadIdx.x;
  const char* g = (const char*)gsrc + tid * 16;
  char* l = (char*)lds + tid * 16;
  // ISA 08_async_tensor: INST_OFFSET is added to both global and LDS addresses
  __builtin_amdgcn_global_load_async_to_lds_b128((g1_v4i)g, (l3_v4i)l, 0, 0);
  __builtin_amdgcn_global_load_async_to_lds_b128((g1_v4i)g, (l3_v4i)l, 2048, 0);
  __builtin_amdgcn_global_load_async_to_lds_b128((g1_v4i)g, (l3_v4i)l, 4096, 0);
  __builtin_amdgcn_global_load_async_to_lds_b128((g1_v4i)g, (l3_v4i)l, 6144, 0);
#else
  const int tid = threadIdx.x;
  for (int i = tid; i < 64 * 64 / 4; i += 128)
    ((unsigned long long*)lds)[i] = ((const unsigned long long*)gsrc)[i];
#endif
}

// ---- WMMA fragment loaders (contiguous k = i + 16*g order, wave32) ----------

// A matrix 16x32 f16: lane (l&15)=row M, group (l>>4): K = i + 16*g
__device__ __forceinline__ v16h load_frag_A(const _Float16* p, int ld) {
  const int lane = threadIdx.x & 31;
  const int m = lane & 15;
  const int g = lane >> 4;
  v16h a;
#pragma unroll
  for (int i = 0; i < 16; ++i) {
    const int kk = i + (g << 4);
    a[i] = p[m * ld + kk];
  }
  return a;
}

// B matrix 32x16 f16, tile stored [n][k] (n-major): element(k,n) = p[n*ld + k]
__device__ __forceinline__ v16h load_frag_B_nmajor(const _Float16* p, int ld) {
  const int lane = threadIdx.x & 31;
  const int n = lane & 15;
  const int g = lane >> 4;
  v16h b;
#pragma unroll
  for (int i = 0; i < 16; ++i) {
    const int kk = i + (g << 4);
    b[i] = p[n * ld + kk];
  }
  return b;
}

// B matrix 32x16 f16, tile stored [k][n] (k-major): element(k,n) = p[k*ld + n]
__device__ __forceinline__ v16h load_frag_B_kmajor(const _Float16* p, int ld) {
  const int lane = threadIdx.x & 31;
  const int n = lane & 15;
  const int g = lane >> 4;
  v16h b;
#pragma unroll
  for (int i = 0; i < 16; ++i) {
    const int kk = i + (g << 4);
    b[i] = p[kk * ld + n];
  }
  return b;
}

#define PT_M 128
#define PT_N 128
#define PT_K 32

// Stage a 128x32 fp32 panel -> f16 LDS tile.
// Phase 1: 4 x b128 loads in flight; Phase 2: packed cvt + b64 LDS stores.
__device__ __forceinline__ void stage_panel_f32(const float* __restrict__ g,
                                                int rowStride,
                                                _Float16* __restrict__ lds) {
  const int tid = threadIdx.x;  // 0..255
  float4 x[4];
#pragma unroll
  for (int jj = 0; jj < 4; ++jj) {
    const int e = (tid + jj * 256) * 4;
    const int r = e >> 5, c = e & 31;
    x[jj] = *(const float4*)(g + (size_t)r * rowStride + c);
  }
#pragma unroll
  for (int jj = 0; jj < 4; ++jj) {
    const int e = (tid + jj * 256) * 4;
    const int r = e >> 5, c = e & 31;
    v4h hv;
    hv[0] = (_Float16)x[jj].x; hv[1] = (_Float16)x[jj].y;
    hv[2] = (_Float16)x[jj].z; hv[3] = (_Float16)x[jj].w;
    *(v4h*)&lds[r * PT_K + c] = hv;
  }
}

// Stage a 128x32 f16 panel -> LDS (straight 8-byte copies, batched).
__device__ __forceinline__ void stage_panel_f16(const _Float16* __restrict__ g,
                                                int rowStride,
                                                _Float16* __restrict__ lds) {
  const int tid = threadIdx.x;
  unsigned long long x[4];
#pragma unroll
  for (int jj = 0; jj < 4; ++jj) {
    const int e = (tid + jj * 256) * 4;
    const int r = e >> 5, c = e & 31;
    x[jj] = *(const unsigned long long*)(g + (size_t)r * rowStride + c);
  }
#pragma unroll
  for (int jj = 0; jj < 4; ++jj) {
    const int e = (tid + jj * 256) * 4;
    const int r = e >> 5, c = e & 31;
    *(unsigned long long*)&lds[r * PT_K + c] = x[jj];
  }
}

// ---- Input projection GEMM: Y = X @ W^T + bias, written per-head f16 -------
// X: [8192][512] fp32, W: [512][512] fp32 (out,in), Y: [(b*8+h)][nRows][64] f16
__global__ __launch_bounds__(256)
void proj_in_kernel(const float* __restrict__ X, const float* __restrict__ W,
                    const float* __restrict__ bias, _Float16* __restrict__ Y,
                    int nRows, float scaleOut)
{
  __shared__ __align__(16) _Float16 As[PT_M * PT_K];
  __shared__ __align__(16) _Float16 Bs[PT_N * PT_K]; // stored [n][k]
  const int K = 512;
  const int bm = blockIdx.x * PT_M;
  const int bn = blockIdx.y * PT_N;
  const int tid = threadIdx.x;
  const int wave = tid >> 5;
  const int lane = tid & 31;
  const int wm = (wave >> 2) * 64;   // 0 or 64
  const int wn = (wave & 3) * 32;    // 0..96
  v8f acc[4][2] = {};

  for (int k0 = 0; k0 < K; k0 += PT_K) {
    stage_panel_f32(&X[(size_t)bm * K + k0], K, As);
    stage_panel_f32(&W[(size_t)bn * K + k0], K, Bs);
    // cover latency of next panel (global_prefetch_b8)
    if (k0 + PT_K < K && tid < 128) {
      __builtin_prefetch(&X[(size_t)(bm + tid) * K + k0 + PT_K], 0, 0);
      __builtin_prefetch(&W[(size_t)(bn + tid) * K + k0 + PT_K], 0, 0);
    }
    __syncthreads();
    const v16h bf0 = load_frag_B_nmajor(&Bs[(wn +  0) * PT_K], PT_K);
    const v16h bf1 = load_frag_B_nmajor(&Bs[(wn + 16) * PT_K], PT_K);
#pragma unroll
    for (int mt = 0; mt < 4; ++mt) {
      const v16h af = load_frag_A(&As[(wm + mt * 16) * PT_K], PT_K);
      acc[mt][0] = WMMA_F32_F16(af, bf0, acc[mt][0]);
      acc[mt][1] = WMMA_F32_F16(af, bf1, acc[mt][1]);
    }
    __syncthreads();
  }

  const int nl = lane & 15, g = lane >> 4;
#pragma unroll
  for (int nt = 0; nt < 2; ++nt) {
    const int ncol = bn + wn + nt * 16 + nl;
    const float bval = bias[ncol];
    const int h = ncol >> 6, d = ncol & 63;
#pragma unroll
    for (int mt = 0; mt < 4; ++mt) {
#pragma unroll
      for (int r = 0; r < 8; ++r) {
        const int m = bm + wm + mt * 16 + r + 8 * g;
        const int batch = m >> 11, row = m & 2047;
        const float val = (acc[mt][nt][r] + bval) * scaleOut;
        Y[((size_t)(batch * 8 + h) * nRows + row) * 64 + d] = (_Float16)val;
      }
    }
  }
}

// ---- Output projection GEMM: Y = A(f16) @ Wo^T + bo, fp32 out --------------
__global__ __launch_bounds__(256)
void proj_out_kernel(const _Float16* __restrict__ Xh, const float* __restrict__ W,
                     const float* __restrict__ bias, float* __restrict__ Y)
{
  __shared__ __align__(16) _Float16 As[PT_M * PT_K];
  __shared__ __align__(16) _Float16 Bs[PT_N * PT_K];
  const int K = 512;
  const int bm = blockIdx.x * PT_M;
  const int bn = blockIdx.y * PT_N;
  const int tid = threadIdx.x;
  const int wave = tid >> 5;
  const int lane = tid & 31;
  const int wm = (wave >> 2) * 64;
  const int wn = (wave & 3) * 32;
  v8f acc[4][2] = {};

  for (int k0 = 0; k0 < K; k0 += PT_K) {
    stage_panel_f16(&Xh[(size_t)bm * K + k0], K, As);
    stage_panel_f32(&W[(size_t)bn * K + k0], K, Bs);
    if (k0 + PT_K < K && tid < 128) {
      __builtin_prefetch(&Xh[(size_t)(bm + tid) * K + k0 + PT_K], 0, 0);
      __builtin_prefetch(&W[(size_t)(bn + tid) * K + k0 + PT_K], 0, 0);
    }
    __syncthreads();
    const v16h bf0 = load_frag_B_nmajor(&Bs[(wn +  0) * PT_K], PT_K);
    const v16h bf1 = load_frag_B_nmajor(&Bs[(wn + 16) * PT_K], PT_K);
#pragma unroll
    for (int mt = 0; mt < 4; ++mt) {
      const v16h af = load_frag_A(&As[(wm + mt * 16) * PT_K], PT_K);
      acc[mt][0] = WMMA_F32_F16(af, bf0, acc[mt][0]);
      acc[mt][1] = WMMA_F32_F16(af, bf1, acc[mt][1]);
    }
    __syncthreads();
  }

  const int nl = lane & 15, g = lane >> 4;
#pragma unroll
  for (int nt = 0; nt < 2; ++nt) {
    const int ncol = bn + wn + nt * 16 + nl;
    const float bval = bias[ncol];
#pragma unroll
    for (int mt = 0; mt < 4; ++mt) {
#pragma unroll
      for (int r = 0; r < 8; ++r) {
        const int m = bm + wm + mt * 16 + r + 8 * g;
        Y[(size_t)m * 512 + ncol] = acc[mt][nt][r] + bval;
      }
    }
  }
}

// ---- Memory-slot rows into kc/vc (rows 2048..2111), scaled by 8 ------------
__global__ __launch_bounds__(256)
void memrows_kernel(const float* __restrict__ m_k, const float* __restrict__ m_v,
                    _Float16* __restrict__ kc, _Float16* __restrict__ vc)
{
  const int idx = blockIdx.x * 256 + threadIdx.x;
  if (idx >= 4 * 8 * 64 * 64) return;
  const int d = idx & 63;
  const int j = (idx >> 6) & 63;
  const int h = (idx >> 12) & 7;
  const int b = idx >> 15;
  const float kv = m_k[j * 512 + h * 64 + d] * 8.0f;  // * sqrt(D_K)
  const float vv = m_v[j * 512 + h * 64 + d] * 8.0f;  // * sqrt(N_MEM)
  const size_t off = (((size_t)(b * 8 + h)) * 2112 + 2048 + j) * 64 + d;
  kc[off] = (_Float16)kv;
  vc[off] = (_Float16)vv;
}

// ---- Flash attention: per (bh, q-tile of 64), 4 waves, online softmax ------
// K/V tiles double-buffered and streamed with GLOBAL_LOAD_ASYNC_TO_LDS_B128:
// tile t+1 is in flight (ASYNCcnt) while tile t feeds 16 WMMAs + softmax.
__global__ __launch_bounds__(128)
void flash_kernel(const _Float16* __restrict__ qh, const _Float16* __restrict__ kc,
                  const _Float16* __restrict__ vc, const int* __restrict__ amask,
                  const float* __restrict__ aw, _Float16* __restrict__ attnOut)
{
  const int NK = 2048, NTOT = 2112, DK = 64, NQ = 2048;
  const int bh = blockIdx.x;          // b*8 + h
  const int b  = bh >> 3;
  const int h  = bh & 7;
  const int qt = blockIdx.y;          // 0..31

  __shared__ __align__(16) _Float16 Qs[64 * 64];
  __shared__ __align__(16) _Float16 Ks[2][64 * 64];
  __shared__ __align__(16) _Float16 Vs[2][64 * 64];
  __shared__ __align__(16) _Float16 Ps[4][16 * 64];

  const int tid = threadIdx.x;
  const int wave = tid >> 5;
  const int lane = tid & 31;
  const int nl = lane & 15, g = lane >> 4;

  // Prologue: start Q tile + K/V tile 0 (12 async b128 per thread in flight)
  const _Float16* qbase  = qh + ((size_t)bh * NQ + (size_t)qt * 64) * DK;
  const _Float16* kvbase = kc + (size_t)bh * NTOT * DK;
  const _Float16* vvbase = vc + (size_t)bh * NTOT * DK;
  tile_copy_8k(qbase, Qs);
  tile_copy_8k(kvbase, Ks[0]);
  tile_copy_8k(vvbase, Vs[0]);

  float mrow[8], lrow[8];
  v8f oacc[4] = {};
#pragma unroll
  for (int r = 0; r < 8; ++r) { mrow[r] = -1e30f; lrow[r] = 0.0f; }

  for (int kt = 0; kt < 33; ++kt) {
    const int cur = kt & 1, nxt = cur ^ 1;
    __syncthreads();  // all waves finished reading buffer `nxt` (iter kt-1)
    if (kt + 1 < 33) {
      tile_copy_8k(kvbase + (size_t)(kt + 1) * 64 * DK, Ks[nxt]);
      tile_copy_8k(vvbase + (size_t)(kt + 1) * 64 * DK, Vs[nxt]);
      WAIT_ASYNC(8);  // 8 newer loads may stay in flight; tile kt has retired
    } else {
      WAIT_ASYNC(0);
    }
    __syncthreads();  // tile kt visible to all waves

    // S(16x64) = Qrows @ K^T : 4 col-tiles x 2 k-steps of 32
    v8f s[4] = {};
#pragma unroll
    for (int ks = 0; ks < 2; ++ks) {
      const v16h af = load_frag_A(&Qs[(wave * 16) * 64 + ks * 32], 64);
#pragma unroll
      for (int nt = 0; nt < 4; ++nt) {
        const v16h bf = load_frag_B_nmajor(&Ks[cur][(nt * 16) * 64 + ks * 32], 64);
        s[nt] = WMMA_F32_F16(af, bf, s[nt]);
      }
    }

    // weights / mask on the non-memory columns (one column per lane per tile)
#pragma unroll
    for (int nt = 0; nt < 4; ++nt) {
      const int cg = kt * 64 + nt * 16 + nl;
      if (cg < NK) {
        const float wgt = aw[b * NK + cg];
        const int   msk = amask[b * NK + cg];
#pragma unroll
        for (int r = 0; r < 8; ++r) {
          const float sv = s[nt][r] * wgt;
          s[nt][r] = msk ? -1e30f : sv;
        }
      }
    }

    // row max across 64 columns (4 in-lane + 16-lane half-wave xor reduce)
    float rmax[8];
#pragma unroll
    for (int r = 0; r < 8; ++r) {
      float v = fmaxf(fmaxf(s[0][r], s[1][r]), fmaxf(s[2][r], s[3][r]));
      v = fmaxf(v, __shfl_xor(v, 1, 32));
      v = fmaxf(v, __shfl_xor(v, 2, 32));
      v = fmaxf(v, __shfl_xor(v, 4, 32));
      v = fmaxf(v, __shfl_xor(v, 8, 32));
      rmax[r] = v;
    }

    // online softmax update
    float corr[8];
#pragma unroll
    for (int r = 0; r < 8; ++r) {
      const float mnew = fmaxf(mrow[r], rmax[r]);
      corr[r] = __expf(mrow[r] - mnew);
      mrow[r] = mnew;
    }
#pragma unroll
    for (int r = 0; r < 8; ++r) {
      float acc = 0.0f;
#pragma unroll
      for (int nt = 0; nt < 4; ++nt) {
        const float p = __expf(s[nt][r] - mrow[r]);
        s[nt][r] = p;
        acc += p;
      }
      acc += __shfl_xor(acc, 1, 32);
      acc += __shfl_xor(acc, 2, 32);
      acc += __shfl_xor(acc, 4, 32);
      acc += __shfl_xor(acc, 8, 32);
      lrow[r] = lrow[r] * corr[r] + acc;
    }
#pragma unroll
    for (int dt = 0; dt < 4; ++dt)
#pragma unroll
      for (int r = 0; r < 8; ++r)
        oacc[dt][r] *= corr[r];

    // P (D-layout) -> per-wave LDS (A-layout source), then O += P @ V
    _Float16* Pw = Ps[wave];
#pragma unroll
    for (int nt = 0; nt < 4; ++nt)
#pragma unroll
      for (int r = 0; r < 8; ++r)
        Pw[(r + 8 * g) * 64 + nt * 16 + nl] = (_Float16)s[nt][r];

#pragma unroll
    for (int ks = 0; ks < 2; ++ks) {
      const v16h af = load_frag_A(&Pw[ks * 32], 64);
#pragma unroll
      for (int dt = 0; dt < 4; ++dt) {
        const v16h bf = load_frag_B_kmajor(&Vs[cur][(ks * 32) * 64 + dt * 16], 64);
        oacc[dt] = WMMA_F32_F16(af, bf, oacc[dt]);
      }
    }
  }

  // normalize and store into [b][row][h*64+d] f16 for the output GEMM
#pragma unroll
  for (int dt = 0; dt < 4; ++dt) {
#pragma unroll
    for (int r = 0; r < 8; ++r) {
      const int qrow = qt * 64 + wave * 16 + r + 8 * g;
      const float o = oacc[dt][r] / lrow[r];
      attnOut[((size_t)b * NQ + qrow) * 512 + h * 64 + dt * 16 + nl] = (_Float16)o;
    }
  }
}

// ---------------------------------------------------------------------------
extern "C" void kernel_launch(void* const* d_in, const int* in_sizes, int n_in,
                              void* d_out, int out_size, void* d_ws, size_t ws_size,
                              hipStream_t stream) {
  const float* q  = (const float*)d_in[0];
  const float* k  = (const float*)d_in[1];
  const float* v  = (const float*)d_in[2];
  const int*   am = (const int*)  d_in[3];
  const float* aw = (const float*)d_in[4];
  const float* Wq = (const float*)d_in[5];  const float* bq = (const float*)d_in[6];
  const float* Wk = (const float*)d_in[7];  const float* bk = (const float*)d_in[8];
  const float* Wv = (const float*)d_in[9];  const float* bv = (const float*)d_in[10];
  const float* Wo = (const float*)d_in[11]; const float* bo = (const float*)d_in[12];
  const float* mk = (const float*)d_in[13]; const float* mv = (const float*)d_in[14];

  char* ws = (char*)d_ws;
  size_t off = 0;
  _Float16* qh = (_Float16*)(ws + off); off += (size_t)32 * 2048 * 64 * 2;  // 8 MB
  _Float16* kc = (_Float16*)(ws + off); off += (size_t)32 * 2112 * 64 * 2;  // 8.25 MB
  _Float16* vc = (_Float16*)(ws + off); off += (size_t)32 * 2112 * 64 * 2;  // 8.25 MB
  _Float16* ao = (_Float16*)(ws + off);                                     // 8 MB

  dim3 gproj(64, 4);  // M=8192/128, N=512/128
  proj_in_kernel<<<gproj, 256, 0, stream>>>(q, Wq, bq, qh, 2048, 0.125f); // scale folded
  proj_in_kernel<<<gproj, 256, 0, stream>>>(k, Wk, bk, kc, 2112, 1.0f);
  proj_in_kernel<<<gproj, 256, 0, stream>>>(v, Wv, bv, vc, 2112, 1.0f);
  memrows_kernel<<<512, 256, 0, stream>>>(mk, mv, kc, vc);
  flash_kernel<<<dim3(32, 32), 128, 0, stream>>>(qh, kc, vc, am, aw, ao);
  proj_out_kernel<<<gproj, 256, 0, stream>>>(ao, Wo, bo, (float*)d_out);
}
